// HighwayBiLSTM_20126216749959
// MI455X (gfx1250) — matrix operand order repaired and checked by reading, not compiled, not run on hardware
//
#include <hip/hip_runtime.h>
#include <stdint.h>
#include <stddef.h>

// ---------------------------------------------------------------------------
// Highway BiLSTM for MI455X (gfx1250, wave32).
//  - All GEMMs: v_wmma_f32_16x16x32_bf16, f32 accumulation.
//  - Weight panels staged to LDS with the Tensor Data Mover, double-buffered
//    on TENSORcnt (s_wait_tensorcnt), shared by all 8 waves of a block.
//  - Recurrence: 16 workgroups (2 dirs x 8 hidden slices), Wh^T slice resident
//    in LDS (TDM-staged), cell state in LDS, h ping-pong through L2, per-step
//    device-scope barrier + s_cluster_barrier (native path when clustered).
// ---------------------------------------------------------------------------

#define LNUM 3
#define BB   64
#define SS   512
#define DD   1024
#define HH   512
#define HID2 1024
#define G3   1536
#define BS   (BB*SS)          // 32768 rows

typedef __attribute__((ext_vector_type(16))) __bf16 v16bf;
typedef __attribute__((ext_vector_type(8)))  float  v8f;
typedef uint32_t u32x4_t __attribute__((ext_vector_type(4)));

union FragU { u32x4_t q[2]; v16bf v; };

// LDS for the recurrent kernel
#define WHS_HALVES (192*512)                      // Wh^T slice, bf16
#define GBUF_F32   (64*192)                       // raw gate GEMM result
#define CSL_F32    (64*64)                        // cell state slice
#define SMEM_REC   (WHS_HALVES*2 + GBUF_F32*4 + CSL_F32*4)   // 262144 B <= 320KB

// LDS for GEMM kernels: double-buffered B panel [2][128 n][64 k] bf16
#define BPANEL_HALVES (128*64)
#define SMEM_GEMM     (2*BPANEL_HALVES*2)          // 32 KB

__device__ __forceinline__ uint16_t f2bf(float f) {
  union { float f; uint32_t u; } v; v.f = f;
  uint32_t u = v.u;
  return (uint16_t)((u + 0x7FFFu + ((u >> 16) & 1u)) >> 16);   // RNE
}

__device__ __forceinline__ v8f wmma_bf16(v16bf a, v16bf b, v8f c) {
  return __builtin_amdgcn_wmma_f32_16x16x32_bf16(
      /*neg_a=*/false, a, /*neg_b=*/false, b,
      /*c_mod=*/(short)0, c, /*reuse_a=*/false, /*reuse_b=*/false);
}

__device__ __forceinline__ float sigm(float x) { return 1.0f / (1.0f + __expf(-x)); }

#if defined(__has_builtin)
#if __has_builtin(__builtin_amdgcn_tensor_load_to_lds) && __has_builtin(__builtin_amdgcn_s_wait_tensorcnt)
#define HB_TDM 1
#endif
#endif

#ifdef HB_TDM
typedef unsigned int tdm_u32x4 __attribute__((ext_vector_type(4)));
typedef int tdm_i32x4 __attribute__((ext_vector_type(4)));
typedef int tdm_i32x8 __attribute__((ext_vector_type(8)));

// Issue one 2D TDM tile load: th rows x tw bf16 elements, global row stride
// `stride` (elements).  The tile is described as its own (padded) tensor.
__device__ __forceinline__ void tdm_issue_2d(const void* gptr, uint32_t ldsoff,
                                             uint32_t tw, uint32_t th, uint32_t stride) {
  const uint64_t ga = (uint64_t)(uintptr_t)gptr;
  tdm_u32x4 g0;
  g0[0] = 1u;                                               // count=1, user mode
  g0[1] = ldsoff;                                           // lds_addr (bytes)
  g0[2] = (uint32_t)ga;                                     // global_addr[31:0]
  g0[3] = ((uint32_t)(ga >> 32) & 0x01FFFFFFu) | (2u << 30); // addr[56:32] | type=2
  tdm_i32x8 g1;
  g1[0] = 0x00010000;                                       // data_size = 2 bytes
  g1[1] = (int)(tw << 16);                                  // tensor_dim0
  g1[2] = (int)(th << 16);                                  // tensor_dim1
  g1[3] = (int)(tw << 16);                                  // tile_dim0
  g1[4] = (int)th;                                          // tile_dim1
  g1[5] = (int)stride;                                      // tensor_dim0_stride
  g1[6] = 0; g1[7] = 0;                                     // dim1 stride unused (2D)
  tdm_i32x4 z4; z4[0] = z4[1] = z4[2] = z4[3] = 0;
#if __clang_major__ >= 23
  tdm_i32x8 z8; z8[0]=z8[1]=z8[2]=z8[3]=z8[4]=z8[5]=z8[6]=z8[7]=0;
  __builtin_amdgcn_tensor_load_to_lds(g0, g1, z4, z4, z8, 0);
#else
  __builtin_amdgcn_tensor_load_to_lds(g0, g1, z4, z4, 0);
#endif
}
#endif

// --------------------------- conversion kernels ----------------------------

// dst[n*K + k] = bf16(src[k*N + n])   (weights stored N-major so a B-operand
// lane fetches two 8-half K-contiguous runs, and TDM tiles are row-contiguous)
__global__ void k_cvt_transpose(const float* __restrict__ src,
                                uint16_t* __restrict__ dst, int K, int N) {
  size_t total = (size_t)K * N;
  for (size_t i = (size_t)blockIdx.x * blockDim.x + threadIdx.x; i < total;
       i += (size_t)gridDim.x * blockDim.x) {
    size_t n = i / K, k = i % K;
    dst[i] = f2bf(src[k * (size_t)N + n]);
  }
}

__global__ void k_cvt_bf16(const float* __restrict__ src,
                           uint16_t* __restrict__ dst, size_t n) {
  for (size_t i = (size_t)blockIdx.x * blockDim.x + threadIdx.x; i < n;
       i += (size_t)gridDim.x * blockDim.x)
    dst[i] = f2bf(src[i]);
}

// ----------------------- generic WMMA GEMM (+bias) -------------------------
// C[m, n] = sum_k A[row(m), k] * Wt[n, k] + bias[n]
// Block tile 64(M) x 128(N); 8 waves.  The 128xK weight panel streams through
// LDS in 64-wide K chunks, double-buffered via TDM (TENSORcnt pipelining).
// permute==1: m enumerates (s,b) s-major, A row = b*SS + s.
__global__ __launch_bounds__(256)
void k_gemm_wmma(const uint16_t* __restrict__ A, const uint16_t* __restrict__ Wt,
                 const float* __restrict__ bias, float* __restrict__ C,
                 int N, int K, int permute) {
  extern __shared__ __align__(16) uint16_t bsm[];   // [2][128][64] bf16
  const int tid = threadIdx.x, wave = tid >> 5, ln = tid & 31;
  const int lane16 = ln & 15;
  const int khalfA = (ln >> 4) * 8;    // A chunk: halves 0..7 -> K+0/8, 8..15 -> K+16/24
  const int kbaseB = (ln >> 4) * 8;    // B chunk: halves 0..7 -> K+0/8, 8..15 -> K+16/24
  const int m0 = blockIdx.x * 64, n0 = blockIdx.y * 128;
  const int mt = wave & 3, ntb = (wave >> 2) * 4;

  const int m_lane = m0 + mt * 16 + lane16;
  size_t a_row;
  if (permute) { int s = m_lane / BB, b = m_lane % BB; a_row = (size_t)b * SS + s; }
  else         a_row = (size_t)m_lane;
  const uint16_t* Ap = A + a_row * (size_t)K;
  __builtin_prefetch(Ap, 0, 1);        // global_prefetch_b8

  const int nchunks = K >> 6;          // 64-wide K chunks
#ifdef HB_TDM
  if (tid < 32)
    tdm_issue_2d(Wt + (size_t)n0 * K, (uint32_t)(uintptr_t)bsm, 64, 128, (uint32_t)K);
#endif

  v8f acc[4] = {};
  for (int ci = 0; ci < nchunks; ++ci) {
    const int kk0 = ci << 6;
    __syncthreads();                   // reads of buf[(ci+1)&1] (iter ci-1) done
#ifdef HB_TDM
    if (tid < 32) {
      if (ci + 1 < nchunks) {
        tdm_issue_2d(Wt + (size_t)n0 * K + (kk0 + 64),
                     (uint32_t)(uintptr_t)bsm + (uint32_t)(((ci + 1) & 1) * BPANEL_HALVES * 2),
                     64, 128, (uint32_t)K);
        __builtin_amdgcn_s_wait_tensorcnt(1);   // chunk ci landed
      } else {
        __builtin_amdgcn_s_wait_tensorcnt(0);
      }
    }
#else
    for (int i = tid; i < BPANEL_HALVES / 8; i += 256) {
      const int r = (i * 8) >> 6, c = (i * 8) & 63;
      *(u32x4_t*)(bsm + ((ci & 1) * BPANEL_HALVES) + i * 8) =
          *(const u32x4_t*)(Wt + (size_t)(n0 + r) * K + kk0 + c);
    }
#endif
    __syncthreads();                   // buf[ci&1] visible to all waves

    const uint16_t* bt = bsm + (ci & 1) * BPANEL_HALVES;
#pragma unroll
    for (int kk2 = 0; kk2 < 64; kk2 += 32) {
      FragU af;
      af.q[0] = *(const u32x4_t*)(Ap + kk0 + kk2 + khalfA);
      af.q[1] = *(const u32x4_t*)(Ap + kk0 + kk2 + 16 + khalfA);
#pragma unroll
      for (int j = 0; j < 4; ++j) {
        const uint16_t* Bp = bt + (size_t)((ntb + j) * 16 + lane16) * 64 + kk2 + kbaseB;
        FragU bfr;
        bfr.q[0] = *(const u32x4_t*)Bp;           // ds_load_b128
        bfr.q[1] = *(const u32x4_t*)(Bp + 16);
        acc[j] = wmma_bf16(af.v, bfr.v, acc[j]);
      }
    }
  }
#pragma unroll
  for (int j = 0; j < 4; ++j) {
    const int n = n0 + (ntb + j) * 16 + lane16;
    const float bv = bias[n];
#pragma unroll
    for (int r = 0; r < 8; ++r) {
      const int m = m0 + mt * 16 + (ln >> 4) * 8 + r;   // C layout: VGPR r, lane half
      C[(size_t)m * N + n] = acc[j][r] + bv;
    }
  }
}

// ---------------- projection GEMM fused with highway gating ----------------
__global__ __launch_bounds__(256)
void k_proj_highway(const uint16_t* __restrict__ A, const uint16_t* __restrict__ Wt,
                    const float* __restrict__ pbias,
                    const float* __restrict__ hcat32, const float* __restrict__ prev32,
                    float* __restrict__ out32, uint16_t* __restrict__ out16,
                    int N, int K) {
  extern __shared__ __align__(16) uint16_t bsm[];   // [2][128][64] bf16
  const int tid = threadIdx.x, wave = tid >> 5, ln = tid & 31;
  const int lane16 = ln & 15;
  const int khalfA = (ln >> 4) * 8, kbaseB = (ln >> 4) * 8;
  const int m0 = blockIdx.x * 64, n0 = blockIdx.y * 128;
  const int mt = wave & 3, ntb = (wave >> 2) * 4;

  const uint16_t* Ap = A + (size_t)(m0 + mt * 16 + lane16) * K;
  __builtin_prefetch(Ap, 0, 1);

  const int nchunks = K >> 6;
#ifdef HB_TDM
  if (tid < 32)
    tdm_issue_2d(Wt + (size_t)n0 * K, (uint32_t)(uintptr_t)bsm, 64, 128, (uint32_t)K);
#endif

  v8f acc[4] = {};
  for (int ci = 0; ci < nchunks; ++ci) {
    const int kk0 = ci << 6;
    __syncthreads();
#ifdef HB_TDM
    if (tid < 32) {
      if (ci + 1 < nchunks) {
        tdm_issue_2d(Wt + (size_t)n0 * K + (kk0 + 64),
                     (uint32_t)(uintptr_t)bsm + (uint32_t)(((ci + 1) & 1) * BPANEL_HALVES * 2),
                     64, 128, (uint32_t)K);
        __builtin_amdgcn_s_wait_tensorcnt(1);
      } else {
        __builtin_amdgcn_s_wait_tensorcnt(0);
      }
    }
#else
    for (int i = tid; i < BPANEL_HALVES / 8; i += 256) {
      const int r = (i * 8) >> 6, c = (i * 8) & 63;
      *(u32x4_t*)(bsm + ((ci & 1) * BPANEL_HALVES) + i * 8) =
          *(const u32x4_t*)(Wt + (size_t)(n0 + r) * K + kk0 + c);
    }
#endif
    __syncthreads();

    const uint16_t* bt = bsm + (ci & 1) * BPANEL_HALVES;
#pragma unroll
    for (int kk2 = 0; kk2 < 64; kk2 += 32) {
      FragU af;
      af.q[0] = *(const u32x4_t*)(Ap + kk0 + kk2 + khalfA);
      af.q[1] = *(const u32x4_t*)(Ap + kk0 + kk2 + 16 + khalfA);
#pragma unroll
      for (int j = 0; j < 4; ++j) {
        const uint16_t* Bp = bt + (size_t)((ntb + j) * 16 + lane16) * 64 + kk2 + kbaseB;
        FragU bfr;
        bfr.q[0] = *(const u32x4_t*)Bp;
        bfr.q[1] = *(const u32x4_t*)(Bp + 16);
        acc[j] = wmma_bf16(af.v, bfr.v, acc[j]);
      }
    }
  }
#pragma unroll
  for (int j = 0; j < 4; ++j) {
    const int n = n0 + (ntb + j) * 16 + lane16;
    const float bv = pbias[n];
#pragma unroll
    for (int r = 0; r < 8; ++r) {
      const int m = m0 + mt * 16 + (ln >> 4) * 8 + r;
      const size_t idx = (size_t)m * N + n;
      const float gate = sigm(acc[j][r] + bv);
      const float o = gate * hcat32[idx] + (1.0f - gate) * prev32[idx];
      out32[idx] = o;
      out16[idx] = f2bf(o);
    }
  }
}

// ----------------------------- recurrence ----------------------------------

__device__ __forceinline__ void grid_barrier(unsigned* syncp, unsigned target) {
  __threadfence();
  __syncthreads();
  if (threadIdx.x == 0) {
    atomicAdd(syncp, 1u);
    while (__hip_atomic_load(syncp, __ATOMIC_ACQUIRE, __HIP_MEMORY_SCOPE_AGENT) < target)
      __builtin_amdgcn_s_sleep(1);
  }
  // Native CDNA5 cluster split-barrier: real sync when cluster-dispatched
  // (16 WGs == max cluster size), architecturally S_NOP otherwise.
  __builtin_amdgcn_s_cluster_barrier();
  __syncthreads();
  __threadfence();
}

// 16 workgroups: dir = bid>>3, hidden slice u0 = (bid&7)*64.  Each WG owns the
// i/j/o gate columns of its 64 hidden units -> Wh^T slice [192][512] bf16 in
// LDS (TDM-staged once), cell state [64][64] f32 in LDS, h ping-pongs in L2.
__global__ __launch_bounds__(256)
void k_recurrent(const uint16_t* __restrict__ whT,      // [2][G3][HH] (this layer)
                 const float* __restrict__ gx,          // [2][SS][BB][G3]
                 uint16_t* __restrict__ hbuf,           // [2 phase][2 dir][BB][HH]
                 float* __restrict__ hcat32, uint16_t* __restrict__ hcat16,
                 const float* __restrict__ h0f, const float* __restrict__ h0b,
                 const float* __restrict__ c0f, const float* __restrict__ c0b,
                 unsigned* __restrict__ syncp) {
  extern __shared__ __align__(16) uint16_t smem[];
  uint16_t* whs = smem;                                  // [192][512]
  float* gbuf = (float*)(smem + WHS_HALVES);             // [64][192]
  float* csl  = gbuf + GBUF_F32;                         // [64][64]

  const int tid = threadIdx.x, bid = blockIdx.x;
  const int dir = bid >> 3;
  const int u0  = (bid & 7) * 64;
  const uint16_t* whTd = whT + (size_t)dir * G3 * HH;
  const float* c0 = dir ? c0b : c0f;
  const float* h0 = dir ? h0b : h0f;

  // ---- stage Wh^T slice (3 chunks of 64 rows x 512 halves) into LDS ----
#ifdef HB_TDM
  if (tid < 32) {                      // issued once, by wave 0 only
#pragma unroll
    for (int g = 0; g < 3; ++g)
      tdm_issue_2d(whTd + ((size_t)(g * HH + u0)) * HH,
                   (uint32_t)(uintptr_t)whs + (uint32_t)(g * 64 * HH * 2),
                   HH, 64, HH);
    __builtin_amdgcn_s_wait_tensorcnt(0);
  }
#else
  for (int i = tid; i < (3 * 64 * HH) / 8; i += 256) {
    const int half0 = i * 8;
    const int row = half0 >> 9, col = half0 & 511;
    const int g = row >> 6, r = row & 63;
    *(u32x4_t*)(whs + half0) =
        *(const u32x4_t*)(whTd + ((size_t)(g * HH + u0 + r)) * HH + col);
  }
#endif

  // ---- init c slice (LDS) and our slice of h0 (phase 0 buffer) ----
  for (int i = tid; i < 64 * 64; i += 256) {
    const int b = i >> 6, u = i & 63;
    csl[i] = c0[u0 + u];
    hbuf[((size_t)dir * BB + b) * HH + u0 + u] = f2bf(h0[u0 + u]);
  }
  grid_barrier(syncp, 16u);            // everyone's h0 slice (and LDS) ready

  const int wave = tid >> 5, ln = tid & 31, lane16 = ln & 15;
  const int khalfA = (ln >> 4) * 8, kbaseB = (ln >> 4) * 8;
  const int mt = wave & 3;             // 16-batch tile
  const int ntb = (wave >> 2) * 6;     // 6 of 12 ntiles (192 cols) per wave half

  const v8f vzero = {};

  for (int t = 0; t < SS; ++t) {
    const int tg = dir ? (SS - 1 - t) : t;
    const uint16_t* hp = hbuf + ((size_t)((t & 1) * 2 + dir)) * BB * HH;       // h_{t-1}
    uint16_t*       hn = hbuf + ((size_t)(((t + 1) & 1) * 2 + dir)) * BB * HH; // h_t

    // ---- 64x192x512 WMMA GEMM: g_raw = h_{t-1} @ Wh_slice ----
    v8f acc[6];
#pragma unroll
    for (int j = 0; j < 6; ++j) acc[j] = vzero;
    const uint16_t* Ap = hp + (size_t)(mt * 16 + lane16) * HH;
    for (int kk = 0; kk < HH; kk += 32) {
      FragU af;
      af.q[0] = *(const u32x4_t*)(Ap + kk + khalfA);
      af.q[1] = *(const u32x4_t*)(Ap + kk + 16 + khalfA);
#pragma unroll
      for (int j = 0; j < 6; ++j) {
        const int nloc = (ntb + j) * 16 + lane16;
        const uint16_t* Bp = whs + (size_t)nloc * HH + kk + kbaseB;  // ds_load_b128
        FragU bfr;
        bfr.q[0] = *(const u32x4_t*)Bp;
        bfr.q[1] = *(const u32x4_t*)(Bp + 16);
        acc[j] = wmma_bf16(af.v, bfr.v, acc[j]);
      }
    }
#pragma unroll
    for (int j = 0; j < 6; ++j) {
      const int nloc = (ntb + j) * 16 + lane16;
#pragma unroll
      for (int r = 0; r < 8; ++r)
        gbuf[(mt * 16 + (ln >> 4) * 8 + r) * 192 + nloc] = acc[j][r];
    }
    __syncthreads();

    // ---- coupled-gate cell update for 64 batch x 64 hidden units ----
    const float* gxrow = gx + ((size_t)dir * SS + tg) * BB * G3;
    for (int i = tid; i < 64 * 64; i += 256) {
      const int b = i >> 6, u = i & 63;
      const float* gxb = gxrow + (size_t)b * G3 + u0 + u;
      const float il = gbuf[b * 192 + u]       + gxb[0];
      const float jl = gbuf[b * 192 + 64 + u]  + gxb[HH];
      const float ol = gbuf[b * 192 + 128 + u] + gxb[2 * HH];
      const float ig = sigm(il);
      const float c  = (1.0f - ig) * csl[i] + ig * tanhf(jl);
      const float h  = tanhf(c) * sigm(ol);
      csl[i] = c;
      hn[(size_t)b * HH + u0 + u] = f2bf(h);
      const size_t hidx = ((size_t)b * SS + tg) * HID2 + (size_t)dir * HH + u0 + u;
      hcat32[hidx] = h;
      hcat16[hidx] = f2bf(h);
    }
    grid_barrier(syncp, 16u * (unsigned)(t + 2));
  }
}

// ------------------------------- launcher ----------------------------------

extern "C" void kernel_launch(void* const* d_in, const int* in_sizes, int n_in,
                              void* d_out, int out_size, void* d_ws, size_t ws_size,
                              hipStream_t stream) {
  (void)in_sizes; (void)n_in; (void)out_size; (void)ws_size;
  const float* x   = (const float*)d_in[0];
  const float* fwW = (const float*)d_in[1];
  const float* fwb = (const float*)d_in[2];
  const float* bwW = (const float*)d_in[3];
  const float* bwb = (const float*)d_in[4];
  const float* h0f = (const float*)d_in[5];
  const float* c0f = (const float*)d_in[6];
  const float* h0b = (const float*)d_in[7];
  const float* c0b = (const float*)d_in[8];
  const float* pW  = (const float*)d_in[9];
  const float* pb  = (const float*)d_in[10];
  const float* iW  = (const float*)d_in[11];
  const float* ib  = (const float*)d_in[12];

  char* ws = (char*)d_ws;
  auto alloc = [&](size_t bytes) -> char* {
    char* p = ws; ws += (bytes + 255) & ~(size_t)255; return p;
  };
  uint16_t* wx_t   = (uint16_t*)alloc((size_t)LNUM * 2 * G3 * DD * 2);   // Wx^T bf16
  uint16_t* wh_t   = (uint16_t*)alloc((size_t)LNUM * 2 * G3 * HH * 2);   // Wh^T bf16
  uint16_t* pw_t   = (uint16_t*)alloc((size_t)LNUM * HID2 * HID2 * 2);   // proj^T bf16
  uint16_t* ipw_t  = (uint16_t*)alloc((size_t)HID2 * DD * 2);            // in_proj^T
  uint16_t* cur16  = (uint16_t*)alloc((size_t)BS * HID2 * 2);            // layer input bf16
  float*    cur32  = (float*)   alloc((size_t)BS * HID2 * 4);            // layer input f32
  float*    hwy32  = (float*)   alloc((size_t)BS * HID2 * 4);            // x@in_proj
  float*    gxbuf  = (float*)   alloc((size_t)2 * SS * BB * G3 * 4);     // x@Wx + b
  float*    hcat32 = (float*)   alloc((size_t)BS * HID2 * 4);            // [hf|hb] f32
  uint16_t* hcat16 = (uint16_t*)alloc((size_t)BS * HID2 * 2);            // [hf|hb] bf16
  uint16_t* hbuf   = (uint16_t*)alloc((size_t)2 * 2 * BB * HH * 2);      // h ping-pong
  unsigned* syncp  = (unsigned*)alloc(256);

  // ---- one-time conversions: weights transposed to N-major bf16 ----
  for (int l = 0; l < LNUM; ++l) {
    for (int d = 0; d < 2; ++d) {
      const float* W = (d ? bwW : fwW) + (size_t)l * (DD + HH) * G3;
      k_cvt_transpose<<<768, 256, 0, stream>>>(
          W, wx_t + (size_t)(l * 2 + d) * G3 * DD, DD, G3);
      k_cvt_transpose<<<768, 256, 0, stream>>>(
          W + (size_t)DD * G3, wh_t + (size_t)(l * 2 + d) * G3 * HH, HH, G3);
    }
    k_cvt_transpose<<<768, 256, 0, stream>>>(
        pW + (size_t)l * HID2 * HID2, pw_t + (size_t)l * HID2 * HID2, HID2, HID2);
  }
  k_cvt_transpose<<<768, 256, 0, stream>>>(iW, ipw_t, DD, HID2);
  k_cvt_bf16<<<2048, 256, 0, stream>>>(x, cur16, (size_t)BS * HID2);

  for (int l = 0; l < LNUM; ++l) {
    // parallel precompute: Gx[dir][s][b][3H] = x_t @ Wx + b
    dim3 ggx(BS / 64, G3 / 128);
    k_gemm_wmma<<<ggx, 256, SMEM_GEMM, stream>>>(
        cur16, wx_t + (size_t)(2 * l + 0) * G3 * DD, fwb + (size_t)l * G3,
        gxbuf, G3, DD, 1);
    k_gemm_wmma<<<ggx, 256, SMEM_GEMM, stream>>>(
        cur16, wx_t + (size_t)(2 * l + 1) * G3 * DD, bwb + (size_t)l * G3,
        gxbuf + (size_t)SS * BB * G3, G3, DD, 1);

    if (l == 0) {  // highway carry for layer 0: x @ in_proj_W + b (reads cur16==x)
      dim3 gip(BS / 64, HID2 / 128);
      k_gemm_wmma<<<gip, 256, SMEM_GEMM, stream>>>(cur16, ipw_t, ib, hwy32, HID2, DD, 0);
    }

    hipMemsetAsync(syncp, 0, 8, stream);
    k_recurrent<<<16, 256, SMEM_REC, stream>>>(
        wh_t + (size_t)(2 * l) * G3 * HH, gxbuf, hbuf, hcat32, hcat16,
        h0f + (size_t)l * HH, h0b + (size_t)l * HH,
        c0f + (size_t)l * HH, c0b + (size_t)l * HH, syncp);

    const float* prev = (l == 0) ? hwy32 : cur32;
    float* o32 = (l == LNUM - 1) ? (float*)d_out : cur32;
    dim3 gpj(BS / 64, HID2 / 128);
    k_proj_highway<<<gpj, 256, SMEM_GEMM, stream>>>(
        hcat16, pw_t + (size_t)l * HID2 * HID2, pb + (size_t)l * HID2,
        hcat32, prev, o32, cur16, HID2, HID2);
  }
}
